// Attention_26972394619538
// MI455X (gfx1250) — matrix operand order repaired and checked by reading, not compiled, not run on hardware
//
#include <hip/hip_runtime.h>
#include <hip/hip_bf16.h>

typedef __bf16 bf16_t;
typedef __attribute__((ext_vector_type(16))) __bf16 v16bf;
typedef __attribute__((ext_vector_type(8)))  __bf16 v8bf;
typedef __attribute__((ext_vector_type(8)))  float  v8f;

#define B_   2
#define S_   2048
#define H_   2048
#define NH_  16
#define NKV_ 4
#define HD_  128
#define LAT_ 256
#define MROWS_ (B_ * S_)   // 4096

// ---------------------------------------------------------------------------
// WMMA fragment helpers (layouts per CDNA5 ISA 7.12.2, wave32)
// A 16x32 bf16: lane(0-15)=row M, half h=lane>>4; elems 0-7 -> K = h*8+e,
//               elems 8-15 -> K = 16 + h*8 + (e-8).
// B 32x16 bf16: lane n (n=lane&15) holds column N=n; same K order as A.
// C/D 16x16 f32: vgpr r, lane: M = r + 8*(lane>>4), N = lane&15.
// ---------------------------------------------------------------------------
__device__ __forceinline__ v16bf load_a16x32(const bf16_t* p0, int lda) {
  const int lane = threadIdx.x & 31;
  const int r = lane & 15, h = lane >> 4;
  const bf16_t* p = p0 + (size_t)r * lda + h * 8;
  v8bf lo = *(const v8bf*)(p);
  v8bf hi = *(const v8bf*)(p + 16);
  v16bf f;
#pragma unroll
  for (int e = 0; e < 8; ++e) { f[e] = lo[e]; f[e + 8] = hi[e]; }
  return f;
}

__device__ __forceinline__ v16bf load_b32x16(const bf16_t* p0, int ldb) {
  const int lane = threadIdx.x & 31;
  const int n = lane & 15, h = lane >> 4;
  const bf16_t* p = p0 + (size_t)n * ldb + h * 8;
  v8bf a = *(const v8bf*)(p);
  v8bf b = *(const v8bf*)(p + 16);
  v16bf f;
#pragma unroll
  for (int e = 0; e < 8; ++e) { f[e] = a[e]; f[e + 8] = b[e]; }
  return f;
}

__device__ __forceinline__ v8f wmma_bf16(v16bf a, v16bf b, v8f c) {
  return __builtin_amdgcn_wmma_f32_16x16x32_bf16(
      /*neg_a=*/false, a, /*neg_b=*/false, b,
      /*c_mod=*/(short)0, c, /*reuse_a=*/false, /*reuse_b=*/false);
}

// ---------------------------------------------------------------------------
// fp32 -> bf16 conversion (grid-stride)
// ---------------------------------------------------------------------------
__global__ void cvt_f32_bf16(const float* __restrict__ in,
                             bf16_t* __restrict__ out, long long n) {
  long long i = (long long)blockIdx.x * blockDim.x + threadIdx.x;
  long long stride = (long long)gridDim.x * blockDim.x;
  for (; i < n; i += stride) out[i] = (bf16_t)in[i];
}

// ---------------------------------------------------------------------------
// Generic WMMA GEMM:  C[M,N] = A[M,Kd] * Bw[N,Kd]^T   (A,Bw bf16 row-major)
// Block = 128 threads = 4 waves; each wave owns a 16(M) x 64(N) strip.
// Double-buffered k-loop: next k-step's fragments are in flight while the
// current step's 4 WMMAs issue, so loadcnt waits overlap matrix work.
// Output: f32 (optionally + gate*addsrc) OR bf16.
// ---------------------------------------------------------------------------
__global__ void gemm_wmma(const bf16_t* __restrict__ A,
                          const bf16_t* __restrict__ Bw,
                          int lda, int ldb, int Kd,
                          float* __restrict__ Cf, bf16_t* __restrict__ Cb,
                          int ldc,
                          const float* __restrict__ addsrc,
                          const float* __restrict__ gatep) {
  const int wid = threadIdx.x >> 5;
  const int lane = threadIdx.x & 31;
  const int m0 = blockIdx.y * 64 + wid * 16;
  const int n0 = blockIdx.x * 64;
  const bf16_t* Arow = A + (size_t)m0 * lda;

  v8f acc[4] = {};
  v16bf a_cur = load_a16x32(Arow, lda);
  v16bf b_cur[4];
#pragma unroll
  for (int j = 0; j < 4; ++j)
    b_cur[j] = load_b32x16(Bw + (size_t)(n0 + 16 * j) * ldb, ldb);

  int k0 = 0;
  for (; k0 < Kd - 32; k0 += 32) {
    // stream prefetch one step further ahead (global_prefetch_b8)
    __builtin_prefetch(Arow + (size_t)(lane & 15) * lda + k0 + 64, 0, 3);
    __builtin_prefetch(Bw + (size_t)(n0 + lane) * ldb + k0 + 64, 0, 3);
    __builtin_prefetch(Bw + (size_t)(n0 + 32 + lane) * ldb + k0 + 64, 0, 3);
    // issue next k-step's loads before consuming current fragments
    v16bf a_nxt = load_a16x32(Arow + k0 + 32, lda);
    v16bf b_nxt[4];
#pragma unroll
    for (int j = 0; j < 4; ++j)
      b_nxt[j] = load_b32x16(Bw + (size_t)(n0 + 16 * j) * ldb + k0 + 32, ldb);
#pragma unroll
    for (int j = 0; j < 4; ++j) acc[j] = wmma_bf16(a_cur, b_cur[j], acc[j]);
    a_cur = a_nxt;
#pragma unroll
    for (int j = 0; j < 4; ++j) b_cur[j] = b_nxt[j];
  }
#pragma unroll
  for (int j = 0; j < 4; ++j) acc[j] = wmma_bf16(a_cur, b_cur[j], acc[j]);

  const int n = lane & 15, h = lane >> 4;
  const float gate = addsrc ? *gatep : 0.0f;
#pragma unroll
  for (int j = 0; j < 4; ++j) {
#pragma unroll
    for (int r = 0; r < 8; ++r) {
      size_t idx = (size_t)(m0 + r + 8 * h) * ldc + (n0 + 16 * j + n);
      float v = acc[j][r];
      if (Cf) {
        if (addsrc) v += gate * addsrc[idx];
        Cf[idx] = v;
      } else {
        Cb[idx] = (bf16_t)v;
      }
    }
  }
}

// ---------------------------------------------------------------------------
// RoPE + repack:  proj f32 (b,s,nh*HD) -> out bf16 [b][s][head][d]
// ---------------------------------------------------------------------------
__global__ void rope_pack(const float* __restrict__ proj,
                          const float* __restrict__ cosT,
                          const float* __restrict__ sinT,
                          bf16_t* __restrict__ out, int nheads) {
  long long i = (long long)blockIdx.x * blockDim.x + threadIdx.x;
  int d = (int)(i % HD_);
  long long t = i / HD_;
  int head = (int)(t % nheads);
  long long t2 = t / nheads;
  int s = (int)(t2 % S_);
  int b = (int)(t2 / S_);
  size_t base = (size_t)(b * S_ + s) * (nheads * HD_) + head * HD_;
  float x = proj[base + d];
  float other = proj[base + ((d < HD_ / 2) ? d + HD_ / 2 : d - HD_ / 2)];
  float rot = (d < HD_ / 2) ? -other : other;
  float c = cosT[(size_t)s * HD_ + d];
  float sn = sinT[(size_t)s * HD_ + d];
  out[i] = (bf16_t)(x * c + rot * sn);  // i == (((b*S+s)*nh+head)*HD+d)
}

// ---------------------------------------------------------------------------
// V repack + transpose: proj f32 (b,s,NKV*HD) -> Vt bf16 [b][hk][d][s]
// ---------------------------------------------------------------------------
__global__ void v_pack(const float* __restrict__ proj, bf16_t* __restrict__ Vt) {
  long long i = (long long)blockIdx.x * blockDim.x + threadIdx.x;
  int s = (int)(i % S_);
  long long t = i / S_;
  int d = (int)(t % HD_);
  long long t2 = t / HD_;
  int hk = (int)(t2 % NKV_);
  int b = (int)(t2 / NKV_);
  Vt[i] = (bf16_t)proj[(size_t)(b * S_ + s) * (NKV_ * HD_) + hk * HD_ + d];
}

// ---------------------------------------------------------------------------
// Flash attention: one wave per (b, head, 16-query tile).
// QK^T and PV both via bf16 WMMA; online softmax; causal mask analytic.
// V fragments are loaded BEFORE the softmax VALU block so their global-load
// latency hides behind exp/shuffle work; next K chunk is prefetched.
// Single-wave workgroup => __syncthreads lowers to s_nop (safe, cheap).
// ---------------------------------------------------------------------------
__global__ void attn_kernel(const bf16_t* __restrict__ Qb,
                            const bf16_t* __restrict__ Kb,
                            const bf16_t* __restrict__ Vt,
                            bf16_t* __restrict__ ctx) {
  __shared__ __align__(16) bf16_t P[16 * 32];
  const int lane = threadIdx.x & 31;
  const int q0 = blockIdx.x * 16;
  const int hq = blockIdx.y;
  const int b = blockIdx.z;
  const int hk = hq >> 2;  // NH/NKV = 4

  const bf16_t* Qh = Qb + (((size_t)b * NH_ + hq) * S_ + q0) * HD_;
  const bf16_t* Kh = Kb + ((size_t)b * NKV_ + hk) * S_ * HD_;
  const bf16_t* Vh = Vt + ((size_t)b * NKV_ + hk) * HD_ * (size_t)S_;

  v16bf qf[4];
#pragma unroll
  for (int i = 0; i < 4; ++i) qf[i] = load_a16x32(Qh + 32 * i, HD_);

  float mrow[8], lrow[8];
#pragma unroll
  for (int r = 0; r < 8; ++r) { mrow[r] = -3.0e38f; lrow[r] = 0.0f; }
  v8f acc[8] = {};

  const int n = lane & 15, h = lane >> 4;
  const float scale = 0.08838834764831845f;  // 1/sqrt(128)
  const int kend = q0 + 16;

  for (int kc = 0; kc < kend; kc += 32) {
    // ---- scores for 32 keys: two 16x16 C tiles ----
    v8f s0 = {}, s1 = {};
#pragma unroll
    for (int i = 0; i < 4; ++i) {
      s0 = wmma_bf16(qf[i], load_b32x16(Kh + (size_t)kc * HD_ + 32 * i, HD_), s0);
      s1 = wmma_bf16(qf[i], load_b32x16(Kh + (size_t)(kc + 16) * HD_ + 32 * i, HD_), s1);
    }
    // ---- issue this chunk's V loads + next chunk's K prefetch now, so the
    //      softmax VALU block below hides their latency ----
    v16bf vf[8];
#pragma unroll
    for (int j = 0; j < 8; ++j)
      vf[j] = load_b32x16(Vh + (size_t)(16 * j) * S_ + kc, S_);
    if (kc + 32 < kend) {
      __builtin_prefetch(Kh + (size_t)(kc + 32 + lane) * HD_, 0, 3);
      __builtin_prefetch(Kh + (size_t)(kc + 32 + lane) * HD_ + 64, 0, 3);
    }
    // ---- scale, causal mask, online softmax ----
    float mnew[8];
#pragma unroll
    for (int r = 0; r < 8; ++r) {
      int qrow = q0 + r + 8 * h;
      float a = s0[r] * scale; if (kc + n > qrow)      a = -1.0e9f;
      float c = s1[r] * scale; if (kc + 16 + n > qrow) c = -1.0e9f;
      s0[r] = a; s1[r] = c;
      float mx = fmaxf(a, c);
      mx = fmaxf(mx, __shfl_xor(mx, 1, 32));
      mx = fmaxf(mx, __shfl_xor(mx, 2, 32));
      mx = fmaxf(mx, __shfl_xor(mx, 4, 32));
      mx = fmaxf(mx, __shfl_xor(mx, 8, 32));
      mnew[r] = fmaxf(mrow[r], mx);
    }
#pragma unroll
    for (int r = 0; r < 8; ++r) {
      float p0 = __expf(s0[r] - mnew[r]);
      float p1 = __expf(s1[r] - mnew[r]);
      P[(r + 8 * h) * 32 + n] = (bf16_t)p0;
      P[(r + 8 * h) * 32 + 16 + n] = (bf16_t)p1;
      float rs = p0 + p1;
      rs += __shfl_xor(rs, 1, 32);
      rs += __shfl_xor(rs, 2, 32);
      rs += __shfl_xor(rs, 4, 32);
      rs += __shfl_xor(rs, 8, 32);
      float cf = __expf(mrow[r] - mnew[r]);
      lrow[r] = lrow[r] * cf + rs;
      mrow[r] = mnew[r];
#pragma unroll
      for (int j = 0; j < 8; ++j) acc[j][r] *= cf;
    }
    __syncthreads();  // single-wave WG: s_nop; orders LDS write->read
    // ---- PV: P(16x32) x V(32x128) ----
    v16bf pf = load_a16x32(P, 32);
#pragma unroll
    for (int j = 0; j < 8; ++j) acc[j] = wmma_bf16(pf, vf[j], acc[j]);
    __syncthreads();
  }

  // ---- epilogue: normalize and scatter to (b,s,h) bf16 for the Wo GEMM ----
#pragma unroll
  for (int r = 0; r < 8; ++r) {
    float inv = 1.0f / lrow[r];
    size_t row = (size_t)(b * S_ + q0 + r + 8 * h);
#pragma unroll
    for (int j = 0; j < 8; ++j)
      ctx[row * H_ + hq * HD_ + 16 * j + n] = (bf16_t)(acc[j][r] * inv);
  }
}

// ---------------------------------------------------------------------------
// Host-side orchestration
// ---------------------------------------------------------------------------
extern "C" void kernel_launch(void* const* d_in, const int* in_sizes, int n_in,
                              void* d_out, int out_size, void* d_ws, size_t ws_size,
                              hipStream_t stream) {
  (void)in_sizes; (void)n_in; (void)out_size; (void)ws_size;
  const float* hs     = (const float*)d_in[0];
  const float* cosT   = (const float*)d_in[1];
  const float* sinT   = (const float*)d_in[2];
  // d_in[3] = attention_mask (causal, recomputed analytically)
  const float* Wq     = (const float*)d_in[4];
  const float* Wk     = (const float*)d_in[5];
  const float* Wv     = (const float*)d_in[6];
  const float* Wo     = (const float*)d_in[7];
  const float* Wl_in  = (const float*)d_in[8];
  const float* Wl_out = (const float*)d_in[9];
  const float* gate   = (const float*)d_in[10];
  float* out = (float*)d_out;

  const size_t MB = 1ull << 20;
  char* ws = (char*)d_ws;
  bf16_t* hs_b = (bf16_t*)(ws + 0);         // 16 MB
  bf16_t* wbuf = (bf16_t*)(ws + 16 * MB);   //  8 MB (largest weight)
  bf16_t* Qb   = (bf16_t*)(ws + 24 * MB);   // 16 MB
  bf16_t* Kb   = (bf16_t*)(ws + 40 * MB);   //  4 MB
  bf16_t* Vt   = (bf16_t*)(ws + 44 * MB);   //  4 MB
  bf16_t* lat1 = (bf16_t*)(ws + 48 * MB);   //  4 MB
  float*  latf = (float*) (ws + 52 * MB);   // 32 MB
  bf16_t* ctxb = (bf16_t*)(ws + 84 * MB);   // 16 MB
  float*  proj = (float*) (ws + 100 * MB);  // 32 MB scratch

  auto cvt = [&](const float* src, bf16_t* dst, long long nelem) {
    int blocks = (int)((nelem + 255) / 256);
    cvt_f32_bf16<<<blocks, 256, 0, stream>>>(src, dst, nelem);
  };

  // hidden states to bf16
  cvt(hs, hs_b, (long long)B_ * S_ * H_);

  // ---- Q projection + RoPE ----
  cvt(Wq, wbuf, (long long)H_ * H_);
  gemm_wmma<<<dim3(H_ / 64, MROWS_ / 64), 128, 0, stream>>>(
      hs_b, wbuf, H_, H_, H_, proj, nullptr, H_, nullptr, nullptr);
  rope_pack<<<(B_ * S_ * NH_ * HD_) / 256, 256, 0, stream>>>(proj, cosT, sinT, Qb, NH_);

  // ---- K projection + RoPE ----
  cvt(Wk, wbuf, (long long)(NKV_ * HD_) * H_);
  gemm_wmma<<<dim3((NKV_ * HD_) / 64, MROWS_ / 64), 128, 0, stream>>>(
      hs_b, wbuf, H_, H_, H_, proj, nullptr, NKV_ * HD_, nullptr, nullptr);
  rope_pack<<<(B_ * S_ * NKV_ * HD_) / 256, 256, 0, stream>>>(proj, cosT, sinT, Kb, NKV_);

  // ---- V projection + transpose ----
  cvt(Wv, wbuf, (long long)(NKV_ * HD_) * H_);
  gemm_wmma<<<dim3((NKV_ * HD_) / 64, MROWS_ / 64), 128, 0, stream>>>(
      hs_b, wbuf, H_, H_, H_, proj, nullptr, NKV_ * HD_, nullptr, nullptr);
  v_pack<<<(B_ * S_ * NKV_ * HD_) / 256, 256, 0, stream>>>(proj, Vt);

  // ---- latent path: hs @ Wl_in^T @ Wl_out^T ----
  cvt(Wl_in, wbuf, (long long)LAT_ * H_);
  gemm_wmma<<<dim3(LAT_ / 64, MROWS_ / 64), 128, 0, stream>>>(
      hs_b, wbuf, H_, H_, H_, nullptr, lat1, LAT_, nullptr, nullptr);
  cvt(Wl_out, wbuf, (long long)H_ * LAT_);
  gemm_wmma<<<dim3(H_ / 64, MROWS_ / 64), 128, 0, stream>>>(
      lat1, wbuf, LAT_, LAT_, LAT_, latf, nullptr, H_, nullptr, nullptr);

  // ---- flash attention ----
  attn_kernel<<<dim3(S_ / 16, NH_, B_), 32, 0, stream>>>(Qb, Kb, Vt, ctxb);

  // ---- output projection + fused latent add: out = ctx@Wo^T + gate*latent ----
  cvt(Wo, wbuf, (long long)H_ * H_);
  gemm_wmma<<<dim3(H_ / 64, MROWS_ / 64), 128, 0, stream>>>(
      ctxb, wbuf, H_, H_, H_, out, nullptr, H_, latf, gate);
}
